// AttentionLayer_9715216024120
// MI455X (gfx1250) — matrix operand order repaired and checked by reading, not compiled, not run on hardware
//
#include <hip/hip_runtime.h>
#include <hip/hip_bf16.h>
#include <math.h>

typedef __bf16 bf16_t;
typedef __attribute__((ext_vector_type(16))) __bf16 v16bf;
typedef __attribute__((ext_vector_type(8)))  __bf16 v8bf;
typedef __attribute__((ext_vector_type(8)))  float  v8f;

#define WMMA_BF16(a, b, c) \
  __builtin_amdgcn_wmma_f32_16x16x32_bf16(false, (a), false, (b), (short)0, (c), false, false)

static __device__ inline v16bf join16(v8bf lo, v8bf hi) {
  v16bf r;
#pragma unroll
  for (int i = 0; i < 8; ++i) { r[i] = lo[i]; r[i + 8] = hi[i]; }
  return r;
}

static __device__ inline v8f vzero8() {
  v8f z;
#pragma unroll
  for (int i = 0; i < 8; ++i) z[i] = 0.f;
  return z;
}

// gfx1250 has native v_tanh_f32; fall back to libm if builtin absent.
static __device__ inline float fast_tanh(float x) {
#if __has_builtin(__builtin_amdgcn_tanhf)
  return __builtin_amdgcn_tanhf(x);
#else
  return tanhf(x);
#endif
}

// Single v_exp_f32 (base-2 exponential).
static __device__ inline float fast_exp2(float x) {
#if __has_builtin(__builtin_amdgcn_exp2f)
  return __builtin_amdgcn_exp2f(x);
#else
  return exp2f(x);
#endif
}

// ---------------------------------------------------------------------------
// Tiled GEMM with tanh epilogue.
// MODE 0: X = fp32 [4096,1024] states, out = bf16 scattered to [b,h,s,d] (QKV).
// MODE 1: X = bf16 [4096,1024] ctx,   out = fp32 [4096,1024] (final output).
// Block tile 128(M) x 64(N), K-step 32, 256 threads = 8 waves.
// Each wave computes a 16x64 strip: 4 accumulators of v8f via WMMA bf16.
// All four B-fragments are preloaded before the WMMA burst so the DS loads
// can be issued as a clause and overlapped with the matrix pipe.
// ---------------------------------------------------------------------------
template <int MODE>
__global__ __launch_bounds__(256) void proj_gemm(const void* __restrict__ Xv,
                                                 const float* __restrict__ W,
                                                 const float* __restrict__ bias,
                                                 void* __restrict__ outv) {
  constexpr int K = 1024, N = 1024;
  __shared__ bf16_t As[128][32];  // [m][k]
  __shared__ bf16_t Bs[64][32];   // transposed: [n][k]

  const int t    = threadIdx.x;
  const int lane = t & 31;
  const int wave = t >> 5;
  const int nl   = lane & 15;
  const int half = lane >> 4;
  const int m0   = blockIdx.y * 128;
  const int n0   = blockIdx.x * 64;

  v8f acc[4];
#pragma unroll
  for (int j = 0; j < 4; ++j) acc[j] = vzero8();

  const int ar = t >> 1;        // 0..127 A row
  const int ak = (t & 1) * 16;  // 0 or 16
  const int bk = t >> 3;        // 0..31  B k
  const int bn = (t & 7) * 8;   // 0..56  B n chunk

  for (int k0 = 0; k0 < K; k0 += 32) {
    // --- stage A tile (convert to bf16) ---
    if (MODE == 0) {
      const float* src = (const float*)Xv + (size_t)(m0 + ar) * K + k0 + ak;
#pragma unroll
      for (int i = 0; i < 4; ++i) {
        float4 f = *(const float4*)(src + i * 4);
        As[ar][ak + i * 4 + 0] = (bf16_t)f.x;
        As[ar][ak + i * 4 + 1] = (bf16_t)f.y;
        As[ar][ak + i * 4 + 2] = (bf16_t)f.z;
        As[ar][ak + i * 4 + 3] = (bf16_t)f.w;
      }
      if (k0 + 32 < K)
        __builtin_prefetch(src + 32, 0, 0);  // global_prefetch_b8 next k-step
    } else {
      const bf16_t* src = (const bf16_t*)Xv + (size_t)(m0 + ar) * K + k0 + ak;
      *(v8bf*)&As[ar][ak]     = *(const v8bf*)(src);
      *(v8bf*)&As[ar][ak + 8] = *(const v8bf*)(src + 8);
      if (k0 + 32 < K)
        __builtin_prefetch(src + 32, 0, 0);
    }
    // --- stage B tile transposed ---
    {
      const float* wsrc = W + (size_t)(k0 + bk) * N + n0 + bn;
      float4 f0 = *(const float4*)(wsrc);
      float4 f1 = *(const float4*)(wsrc + 4);
      Bs[bn + 0][bk] = (bf16_t)f0.x; Bs[bn + 1][bk] = (bf16_t)f0.y;
      Bs[bn + 2][bk] = (bf16_t)f0.z; Bs[bn + 3][bk] = (bf16_t)f0.w;
      Bs[bn + 4][bk] = (bf16_t)f1.x; Bs[bn + 5][bk] = (bf16_t)f1.y;
      Bs[bn + 6][bk] = (bf16_t)f1.z; Bs[bn + 7][bk] = (bf16_t)f1.w;
      if (k0 + 32 < K)
        __builtin_prefetch(wsrc + (size_t)32 * N, 0, 0);
    }
    __syncthreads();

    // A fragment: lane holds row (wave*16 + nl); k layout per ISA 16-bit A 16x32
    v16bf a = join16(*(const v8bf*)&As[wave * 16 + nl][half * 8],
                     *(const v8bf*)&As[wave * 16 + nl][16 + half * 8]);
    // Preload all B fragments, then burst the WMMAs.
    v16bf bfr[4];
#pragma unroll
    for (int j = 0; j < 4; ++j) {
      const int n = j * 16 + nl;
      bfr[j] = join16(*(const v8bf*)&Bs[n][half * 8],
                      *(const v8bf*)&Bs[n][16 + half * 8]);
    }
#pragma unroll
    for (int j = 0; j < 4; ++j) acc[j] = WMMA_BF16(a, bfr[j], acc[j]);
    __syncthreads();
  }

  // epilogue: tanh(acc + bias)
#pragma unroll
  for (int j = 0; j < 4; ++j) {
    const int ng = n0 + j * 16 + nl;
    const float bv = bias[ng];
#pragma unroll
    for (int r = 0; r < 8; ++r) {
      const int mg = m0 + wave * 16 + r + half * 8;
      const float val = fast_tanh(acc[j][r] + bv);
      if (MODE == 0) {
        const int b  = mg >> 11;
        const int sr = mg & 2047;
        const int hh = ng >> 6;
        const int dd = ng & 63;
        ((bf16_t*)outv)[(((size_t)b * 16 + hh) * 2048 + sr) * 64 + dd] = (bf16_t)val;
      } else {
        ((float*)outv)[(size_t)mg * 1024 + ng] = val;
      }
    }
  }
}

// ---------------------------------------------------------------------------
// Causal flash attention. Q/K/V bf16 [B*H][2048][64].
// Block: 128 threads (4 waves) handles 64 q-rows of one (b,h); each wave 16 rows.
// Softmax runs in the base-2 domain: logits pre-scaled by (1/sqrt(64))*log2(e),
// so every exponential is a single v_exp_f32.
// All K/V fragments of a tile are preloaded into distinct registers before the
// WMMA burst so global/DS loads clause up and overlap the matrix pipe instead
// of serializing on s_wait_loadcnt 0.
// ctx written bf16 row-major [B*S][1024] at column h*64+d.
// ---------------------------------------------------------------------------
__global__ __launch_bounds__(128) void flash_attn(const bf16_t* __restrict__ Q,
                                                  const bf16_t* __restrict__ Kb,
                                                  const bf16_t* __restrict__ Vb,
                                                  bf16_t* __restrict__ ctx) {
  __shared__ bf16_t Pl[4][16][64];  // per-wave P tile (A-fragment restage)
  __shared__ bf16_t Vt[64][64];     // V transposed: [d][kv]

  const int t = threadIdx.x, lane = t & 31, wave = t >> 5;
  const int nl = lane & 15, half = lane >> 4;
  const int qt = blockIdx.x;
  const int bh = blockIdx.y;  // b*16 + h
  const int b = bh >> 4, h = bh & 15;
  const int q0 = qt * 64;
  const int wq0 = q0 + wave * 16;
  const size_t base = (size_t)bh * 2048 * 64;

  const float SCALE_LOG2E = 0.125f * 1.44269504088896340736f;  // 1/sqrt(64)*log2(e)

  // Q fragments kept in registers: chunks c=0,1 cover d=0..31, 32..63
  v16bf qf[2];
#pragma unroll
  for (int c = 0; c < 2; ++c) {
    const bf16_t* qp = Q + base + (size_t)(wq0 + nl) * 64 + c * 32;
    qf[c] = join16(*(const v8bf*)(qp + half * 8),
                   *(const v8bf*)(qp + 16 + half * 8));
  }

  float mrow[8], lrow[8];  // running max / sum, base-2 domain
  v8f o[4];
#pragma unroll
  for (int r = 0; r < 8; ++r) { mrow[r] = -1e30f; lrow[r] = 0.f; }
#pragma unroll
  for (int j = 0; j < 4; ++j) o[j] = vzero8();

  const int nt = qt + 1;  // causal: kv tiles 0..qt (uniform across waves)
  for (int tkv = 0; tkv < nt; ++tkv) {
    const int kv0 = tkv * 64;

    // prefetch next KV tile
    if (tkv + 1 < nt) {
      __builtin_prefetch(Kb + base + (size_t)(kv0 + 64 + lane * 2) * 64, 0, 0);
      __builtin_prefetch(Vb + base + (size_t)(kv0 + 64 + lane * 2) * 64, 0, 0);
    }

    // ---- S = Q @ K^T: preload all 8 K fragments, then burst 8 WMMAs ----
    v16bf bkf[4][2];
#pragma unroll
    for (int j = 0; j < 4; ++j) {
      const bf16_t* kp = Kb + base + (size_t)(kv0 + j * 16 + nl) * 64;
#pragma unroll
      for (int c = 0; c < 2; ++c)
        bkf[j][c] = join16(*(const v8bf*)(kp + c * 32 + half * 8),
                           *(const v8bf*)(kp + c * 32 + 16 + half * 8));
    }
    v8f s[4];
#pragma unroll
    for (int j = 0; j < 4; ++j) s[j] = vzero8();
#pragma unroll
    for (int j = 0; j < 4; ++j)
#pragma unroll
      for (int c = 0; c < 2; ++c) s[j] = WMMA_BF16(qf[c], bkf[j][c], s[j]);

    // ---- scale(log2 domain), causal mask, online softmax ----
#pragma unroll
    for (int r = 0; r < 8; ++r) {
      const int row = wq0 + r + half * 8;
      float sv[4];
#pragma unroll
      for (int j = 0; j < 4; ++j) {
        const int col = kv0 + j * 16 + nl;
        const float x = s[j][r] * SCALE_LOG2E;
        sv[j] = (col > row) ? -1e30f : x;
      }
      float tm = fmaxf(fmaxf(sv[0], sv[1]), fmaxf(sv[2], sv[3]));
      tm = fmaxf(tm, __shfl_xor(tm, 1, 32));
      tm = fmaxf(tm, __shfl_xor(tm, 2, 32));
      tm = fmaxf(tm, __shfl_xor(tm, 4, 32));
      tm = fmaxf(tm, __shfl_xor(tm, 8, 32));
      const float mnew = fmaxf(mrow[r], tm);
      const float corr = fast_exp2(mrow[r] - mnew);
      float ps = 0.f;
#pragma unroll
      for (int j = 0; j < 4; ++j) {
        const float p = fast_exp2(sv[j] - mnew);
        s[j][r] = p;
        ps += p;
      }
      ps += __shfl_xor(ps, 1, 32);
      ps += __shfl_xor(ps, 2, 32);
      ps += __shfl_xor(ps, 4, 32);
      ps += __shfl_xor(ps, 8, 32);
      lrow[r] = lrow[r] * corr + ps;
      mrow[r] = mnew;
#pragma unroll
      for (int j = 0; j < 4; ++j) o[j][r] *= corr;
    }

    __syncthreads();  // previous iteration's Pl/Vt consumers are done

    // ---- stage P (C-layout -> A-layout via LDS) and V transposed ----
#pragma unroll
    for (int r = 0; r < 8; ++r)
#pragma unroll
      for (int j = 0; j < 4; ++j)
        Pl[wave][r + half * 8][j * 16 + nl] = (bf16_t)s[j][r];
    {
      const int rv = t >> 1;        // kv row 0..63
      const int dh = (t & 1) * 32;  // d half
      const bf16_t* vp = Vb + base + (size_t)(kv0 + rv) * 64 + dh;
#pragma unroll
      for (int i = 0; i < 4; ++i) {
        v8bf vv = *(const v8bf*)(vp + i * 8);
#pragma unroll
        for (int e = 0; e < 8; ++e) Vt[dh + i * 8 + e][rv] = vv[e];
      }
    }
    __syncthreads();

    // ---- O += P @ V: preload P and V fragments, then burst 8 WMMAs ----
    v16bf ap[2];
#pragma unroll
    for (int c = 0; c < 2; ++c) {
      const bf16_t* pp = &Pl[wave][nl][c * 32];
      ap[c] = join16(*(const v8bf*)(pp + half * 8),
                     *(const v8bf*)(pp + 16 + half * 8));
    }
    v16bf bvf[4][2];
#pragma unroll
    for (int j = 0; j < 4; ++j)
#pragma unroll
      for (int c = 0; c < 2; ++c) {
        const bf16_t* vtp = &Vt[j * 16 + nl][c * 32];
        bvf[j][c] = join16(*(const v8bf*)(vtp + half * 8),
                           *(const v8bf*)(vtp + 16 + half * 8));
      }
#pragma unroll
    for (int j = 0; j < 4; ++j)
#pragma unroll
      for (int c = 0; c < 2; ++c) o[j] = WMMA_BF16(ap[c], bvf[j][c], o[j]);
  }

  // ---- normalize and write ctx (bf16, row-major [B*S][1024]) ----
#pragma unroll
  for (int j = 0; j < 4; ++j)
#pragma unroll
    for (int r = 0; r < 8; ++r) {
      const int row = wq0 + r + half * 8;
      const float val = o[j][r] / lrow[r];
      ctx[((size_t)b * 2048 + row) * 1024 + h * 64 + j * 16 + nl] = (bf16_t)val;
    }
}

// ---------------------------------------------------------------------------
extern "C" void kernel_launch(void* const* d_in, const int* in_sizes, int n_in,
                              void* d_out, int out_size, void* d_ws, size_t ws_size,
                              hipStream_t stream) {
  (void)in_sizes; (void)n_in; (void)out_size; (void)ws_size;
  const float* states = (const float*)d_in[0];
  const float* Wq = (const float*)d_in[1];
  const float* bq = (const float*)d_in[2];
  const float* Wk = (const float*)d_in[3];
  const float* bk = (const float*)d_in[4];
  const float* Wv = (const float*)d_in[5];
  const float* bv = (const float*)d_in[6];
  const float* Wo = (const float*)d_in[7];
  const float* bo = (const float*)d_in[8];

  // workspace: Q,K,V bf16 [B*H*S*64] each, ctx bf16 [B*S*1024]  (32 MB total)
  const size_t QKV_ELEMS = (size_t)2 * 16 * 2048 * 64;  // 4,194,304
  bf16_t* qb = (bf16_t*)d_ws;
  bf16_t* kb = qb + QKV_ELEMS;
  bf16_t* vb = kb + QKV_ELEMS;
  bf16_t* cb = vb + QKV_ELEMS;

  dim3 pg(16, 32);  // N/64 x M/128   (M=4096, N=1024)
  dim3 pb(256);
  proj_gemm<0><<<pg, pb, 0, stream>>>(states, Wq, bq, qb);
  proj_gemm<0><<<pg, pb, 0, stream>>>(states, Wk, bk, kb);
  proj_gemm<0><<<pg, pb, 0, stream>>>(states, Wv, bv, vb);

  flash_attn<<<dim3(32, 32), dim3(128), 0, stream>>>(qb, kb, vb, cb);

  proj_gemm<1><<<pg, pb, 0, stream>>>(cb, Wo, bo, d_out);
}
